// Str_encoder_78323023610215
// MI455X (gfx1250) — compile-verified, matrix operand order checked by reading, not run on hardware
//
#include <hip/hip_runtime.h>
#include <math.h>

#define N_API  50000
#define N_FILE 20000
#define NEDGE  250000
#define NINP   512
#define NOUT   256
#define NH     3
#define FDIM   768   // NH * NOUT

typedef __attribute__((ext_vector_type(16))) __bf16 v16bf;
typedef __attribute__((ext_vector_type(8)))  float  v8f;

union FragBF { v16bf v; unsigned int u[8]; };

__device__ __forceinline__ unsigned short f2bf(float f) {
    unsigned int u = __float_as_uint(f);
    u += 0x7FFFu + ((u >> 16) & 1u);               // round-to-nearest-even
    return (unsigned short)(u >> 16);
}

// RNE both floats, then merge high halves with one v_perm_b32
__device__ __forceinline__ unsigned int pack2bf(float a, float b) {
    unsigned int ua = __float_as_uint(a);
    unsigned int ub = __float_as_uint(b);
    ua += 0x7FFFu + ((ua >> 16) & 1u);
    ub += 0x7FFFu + ((ub >> 16) & 1u);
    // result bytes [1:0] = ua bytes [3:2], bytes [3:2] = ub bytes [3:2]
    return __builtin_amdgcn_perm(ub, ua, 0x07060302);
}

// float atomic max via sign-split int/uint atomics (works across mixed signs)
__device__ __forceinline__ void atomicMaxFloat(float* addr, float val) {
    if (val >= 0.0f) atomicMax((int*)addr, __float_as_int(val));
    else             atomicMin((unsigned int*)addr, __float_as_uint(val));
}

__device__ __forceinline__ float waveReduceSum(float v) {
    for (int off = 16; off > 0; off >>= 1) v += __shfl_xor(v, off, 32);
    return v;
}

// ---------------------------------------------------------------------------
// Weight pre-pack: W f32 [K x N] row-major  ->  Bt bf16 [N x K]
// ---------------------------------------------------------------------------
__global__ void pack_bt_kernel(const float* __restrict__ W,
                               unsigned short* __restrict__ Bt, int K, int N) {
    int idx = blockIdx.x * blockDim.x + threadIdx.x;
    if (idx >= K * N) return;
    int k = idx / N, n = idx - k * N;
    Bt[(size_t)n * K + k] = f2bf(W[(size_t)k * N + n]);
}

// ---------------------------------------------------------------------------
// bf16 WMMA GEMM: C[MxN] = A[MxK] * Bt^T + bias
//   ABF16: A already bf16 row-major; else f32 (converted on LDS store).
//   OBF16: store C as bf16; else f32.
//   Bt: bf16 [N x K].  N multiple of 128, K multiple of 32.
//   Block tile 64x128, 8 wave32 waves (4x2); each wave 16x64 C strip
//   = 4 v_wmma_f32_16x16x32_bf16 per K-step off one A fragment.
//   Software pipelined: next global tile fetched during WMMA phase.
// ---------------------------------------------------------------------------
template<bool ABF16, bool OBF16>
__global__ __launch_bounds__(256) void gemm_wmma(
        const void* __restrict__ Av, const unsigned short* __restrict__ Bt,
        const float* __restrict__ bias, void* __restrict__ Cv,
        int M, int N, int K) {
    const float*          Af = (const float*)Av;
    const unsigned short* Ah = (const unsigned short*)Av;
    float*          Cf = (float*)Cv;
    unsigned short* Ch = (unsigned short*)Cv;

    __shared__ unsigned int lds_a[64][16];    // 64 rows  x 32 k (bf16 pairs)
    __shared__ unsigned int lds_b[128][16];   // 128 cols x 32 k (bf16 pairs)

    const int tid  = threadIdx.x;
    const int lane = tid & 31;
    const int wave = tid >> 5;
    const int hlf  = lane >> 4;
    const int l16  = lane & 15;

    const int bn0 = blockIdx.x * 128;
    const int bm0 = blockIdx.y * 64;

    const int wm = (wave >> 1) * 16;
    const int wn = (wave & 1) * 64;

    v8f acc[4] = {{}, {}, {}, {}};

    // A-load indexing
    const int atr = tid >> 3, atc = tid & 7;   // f32 path: rows atr, atr+32
    const int abr = tid >> 2, abc = tid & 3;   // bf16 path: row abr, uint4 abc
    // B-load indexing
    const int bnr = tid >> 1, bcc = tid & 1;

    float4 ar0, ar1;       // f32-A staging
    uint4  arb;            // bf16-A staging
    uint4  br0, br1;       // B staging

    auto fetch = [&](int k0) {
        if constexpr (ABF16) {
            arb = make_uint4(0u, 0u, 0u, 0u);
            int grow = bm0 + abr;
            if (grow < M)
                arb = ((const uint4*)(Ah + (size_t)grow * K + k0))[abc];
        } else {
            ar0 = make_float4(0.f, 0.f, 0.f, 0.f);
            ar1 = make_float4(0.f, 0.f, 0.f, 0.f);
            int g0 = bm0 + atr, g1 = bm0 + atr + 32;
            if (g0 < M) ar0 = *(const float4*)(Af + (size_t)g0 * K + k0 + atc * 4);
            if (g1 < M) ar1 = *(const float4*)(Af + (size_t)g1 * K + k0 + atc * 4);
        }
        const uint4* bp = (const uint4*)(Bt + (size_t)(bn0 + bnr) * K + k0);
        br0 = bp[bcc * 2 + 0];
        br1 = bp[bcc * 2 + 1];
    };

    auto stage = [&]() {
        if constexpr (ABF16) {
            lds_a[abr][abc * 4 + 0] = arb.x;
            lds_a[abr][abc * 4 + 1] = arb.y;
            lds_a[abr][abc * 4 + 2] = arb.z;
            lds_a[abr][abc * 4 + 3] = arb.w;
        } else {
            lds_a[atr][atc * 2 + 0]      = pack2bf(ar0.x, ar0.y);
            lds_a[atr][atc * 2 + 1]      = pack2bf(ar0.z, ar0.w);
            lds_a[atr + 32][atc * 2 + 0] = pack2bf(ar1.x, ar1.y);
            lds_a[atr + 32][atc * 2 + 1] = pack2bf(ar1.z, ar1.w);
        }
        lds_b[bnr][bcc * 8 + 0] = br0.x;
        lds_b[bnr][bcc * 8 + 1] = br0.y;
        lds_b[bnr][bcc * 8 + 2] = br0.z;
        lds_b[bnr][bcc * 8 + 3] = br0.w;
        lds_b[bnr][bcc * 8 + 4] = br1.x;
        lds_b[bnr][bcc * 8 + 5] = br1.y;
        lds_b[bnr][bcc * 8 + 6] = br1.z;
        lds_b[bnr][bcc * 8 + 7] = br1.w;
    };

    fetch(0);
    for (int k0 = 0; k0 < K; k0 += 32) {
        stage();
        __syncthreads();
        if (k0 + 32 < K) fetch(k0 + 32);   // overlap HBM with WMMA phase

        FragBF afr, bfr[4];
#pragma unroll
        for (int i = 0; i < 8; ++i) {
            int kp = ((((i >> 2) << 4) + (hlf << 3) + ((i & 3) << 1)) >> 1);
            afr.u[i] = lds_a[wm + l16][kp];
#pragma unroll
            for (int t = 0; t < 4; ++t)
                bfr[t].u[i] = lds_b[wn + t * 16 + l16][kp];
        }
#pragma unroll
        for (int t = 0; t < 4; ++t)
            acc[t] = __builtin_amdgcn_wmma_f32_16x16x32_bf16(
                         false, afr.v, false, bfr[t].v, (short)0, acc[t],
                         false, false);
        __syncthreads();
    }

    // epilogue: VGPR r -> row wm + 8*hlf + r; col = wn + 16t + l16
#pragma unroll
    for (int r = 0; r < 8; ++r) {
        int grow = bm0 + wm + hlf * 8 + r;
        if (grow < M) {
#pragma unroll
            for (int t = 0; t < 4; ++t) {
                int gc = bn0 + wn + t * 16 + l16;
                float v = acc[t][r] + (bias ? bias[gc] : 0.f);
                if constexpr (OBF16) Ch[(size_t)grow * N + gc] = f2bf(v);
                else                 Cf[(size_t)grow * N + gc] = v;
            }
        }
    }
}

// ---------------------------------------------------------------------------
// out[n,h] = dot(F[n, h*256 : (h+1)*256], attn[h, :])   (one wave per (n,h))
// ---------------------------------------------------------------------------
__global__ void headdot768_kernel(const float* __restrict__ F,
                                  const float* __restrict__ attn,
                                  float* __restrict__ out, int nrows) {
    int gw   = blockIdx.x * 8 + (threadIdx.x >> 5);
    int lane = threadIdx.x & 31;
    if (gw >= nrows * NH) return;
    int n = gw / NH, h = gw - n * NH;
    const float* fr = F + (size_t)n * FDIM + h * NOUT;
    const float* ar = attn + h * NOUT;
    float p = 0.f;
    for (int d = lane; d < NOUT; d += 32) p += fr[d] * ar[d];
    p = waveReduceSum(p);
    if (lane == 0) out[n * NH + h] = p;
}

// er[n,h] = sum_k Xfile[n,k] * wr[k,h]     (one wave per (n,h))
__global__ void er_kernel(const float* __restrict__ X,
                          const float* __restrict__ wr,
                          float* __restrict__ er, int nrows) {
    int gw   = blockIdx.x * 8 + (threadIdx.x >> 5);
    int lane = threadIdx.x & 31;
    if (gw >= nrows * NH) return;
    int n = gw / NH, h = gw - n * NH;
    float p = 0.f;
    for (int k = lane; k < NINP; k += 32) p += X[(size_t)n * NINP + k] * wr[k * NH + h];
    p = waveReduceSum(p);
    if (lane == 0) er[n * NH + h] = p;
}

__global__ void init_mz_kernel(float* __restrict__ m, float* __restrict__ z, int n) {
    int i = blockIdx.x * blockDim.x + threadIdx.x;
    if (i < n) { m[i] = -INFINITY; z[i] = 0.f; }
}

__global__ void rst_init_kernel(float* __restrict__ rst,
                                const float* __restrict__ gat_bias, int total) {
    int i = blockIdx.x * blockDim.x + threadIdx.x;
    if (i < total) rst[i] = gat_bias[i % FDIM];
}

// e = leaky_relu(el[src]+er[dst], 0.2); segment max into m
__global__ void edge_e_kernel(const int* __restrict__ src, const int* __restrict__ dst,
                              const float* __restrict__ el, const float* __restrict__ er,
                              float* __restrict__ ebuf, float* __restrict__ m) {
    int idx = blockIdx.x * blockDim.x + threadIdx.x;
    if (idx >= NEDGE * NH) return;
    int i = idx / NH, h = idx - i * NH;
    int s = src[i], d = dst[i];
    float v = el[s * NH + h] + er[d * NH + h];
    v = v > 0.f ? v : 0.2f * v;
    ebuf[idx] = v;
    atomicMaxFloat(&m[d * NH + h], v);
}

// a = exp(e - m[dst]); segment sum into z
__global__ void edge_a_kernel(const int* __restrict__ dst,
                              float* __restrict__ ebuf,
                              const float* __restrict__ m, float* __restrict__ z) {
    int idx = blockIdx.x * blockDim.x + threadIdx.x;
    if (idx >= NEDGE * NH) return;
    int i = idx / NH, h = idx - i * NH;
    int d = dst[i];
    float a = expf(ebuf[idx] - m[d * NH + h]);
    ebuf[idx] = a;
    atomicAdd(&z[d * NH + h], a);
}

// rst[dst] += (a/z[dst]) * Fsrc[src]   -- one block per edge, 768 elems
__global__ __launch_bounds__(256) void agg_kernel(
        const int* __restrict__ src, const int* __restrict__ dst,
        const float* __restrict__ ebuf, const float* __restrict__ z,
        const float* __restrict__ Fsrc, float* __restrict__ rst) {
    int e = blockIdx.x;
    int s = src[e], d = dst[e];
#pragma unroll
    for (int j = 0; j < NH; ++j) {   // col in [j*256, (j+1)*256) -> head j
        float alpha = ebuf[e * NH + j] / z[d * NH + j];
        int col = threadIdx.x + j * 256;
        atomicAdd(&rst[(size_t)d * FDIM + col],
                  alpha * Fsrc[(size_t)s * FDIM + col]);
    }
}

// ---------------------------------------------------------------------------
extern "C" void kernel_launch(void* const* d_in, const int* in_sizes, int n_in,
                              void* d_out, int out_size, void* d_ws, size_t ws_size,
                              hipStream_t stream) {
    const float* emb_api  = (const float*)d_in[0];
    const float* emb_file = (const float*)d_in[1];
    // d_in[2] = e_tensor (unused by reference)
    const int*   src      = (const int*)d_in[3];
    const int*   dst      = (const int*)d_in[4];
    const float* Wapi     = (const float*)d_in[5];
    const float* bapi     = (const float*)d_in[6];
    const float* Wfile    = (const float*)d_in[7];
    const float* bfile    = (const float*)d_in[8];
    const float* Wsrc     = (const float*)d_in[9];
    const float* Wdst     = (const float*)d_in[10];
    const float* attn_l   = (const float*)d_in[11];
    const float* attn_r   = (const float*)d_in[12];
    const float* gat_bias = (const float*)d_in[13];
    const float* Whead    = (const float*)d_in[14];
    const float* bhead    = (const float*)d_in[15];
    float* out = (float*)d_out;

    // workspace carving (256B aligned)
    size_t off = 0;
    auto carve = [&](size_t bytes) -> void* {
        void* p = (char*)d_ws + off;
        off = (off + bytes + 255) & ~(size_t)255;
        return p;
    };
    unsigned short* XapiB = (unsigned short*)carve((size_t)N_API * NINP * 2);
    float* Xfile = (float*)carve((size_t)N_FILE * NINP * 4);
    float* Fsrc  = (float*)carve((size_t)N_API  * FDIM * 4);
    float* rst   = (float*)carve((size_t)N_FILE * FDIM * 4);
    float* el    = (float*)carve((size_t)N_API  * NH * 4);
    float* er    = (float*)carve((size_t)N_FILE * NH * 4);
    float* mbuf  = (float*)carve((size_t)N_FILE * NH * 4);
    float* zbuf  = (float*)carve((size_t)N_FILE * NH * 4);
    float* ebuf  = (float*)carve((size_t)NEDGE  * NH * 4);
    float* wr    = (float*)carve((size_t)NINP   * NH * 4);
    unsigned short* WapiT  = (unsigned short*)carve((size_t)NINP * NINP * 2);
    unsigned short* WfileT = (unsigned short*)carve((size_t)NINP * NINP * 2);
    unsigned short* WsrcT  = (unsigned short*)carve((size_t)NINP * FDIM * 2);
    unsigned short* WheadT = (unsigned short*)carve((size_t)FDIM * NOUT * 2);

    // 1) pack weights to bf16 [N x K]
    pack_bt_kernel<<<(NINP * NINP + 255) / 256, 256, 0, stream>>>(Wapi,  WapiT,  NINP, NINP);
    pack_bt_kernel<<<(NINP * NINP + 255) / 256, 256, 0, stream>>>(Wfile, WfileT, NINP, NINP);
    pack_bt_kernel<<<(NINP * FDIM + 255) / 256, 256, 0, stream>>>(Wsrc,  WsrcT,  NINP, FDIM);
    pack_bt_kernel<<<(FDIM * NOUT + 255) / 256, 256, 0, stream>>>(Whead, WheadT, FDIM, NOUT);

    // 2) Xapi(bf16) = emb_api @ Wapi + bapi ; Xfile(f32) = emb_file @ Wfile + bfile
    gemm_wmma<false, true><<<dim3(NINP / 128, (N_API + 63) / 64), 256, 0, stream>>>(
        emb_api, WapiT, bapi, XapiB, N_API, NINP, NINP);
    gemm_wmma<false, false><<<dim3(NINP / 128, (N_FILE + 63) / 64), 256, 0, stream>>>(
        emb_file, WfileT, bfile, Xfile, N_FILE, NINP, NINP);

    // 3) Fsrc = Xapi @ Wsrc   (native bf16 A operand)
    gemm_wmma<true, false><<<dim3(FDIM / 128, (N_API + 63) / 64), 256, 0, stream>>>(
        XapiB, WsrcT, nullptr, Fsrc, N_API, FDIM, NINP);

    // 4) el[n,h] = <Fsrc[n,h,:], attn_l[h,:]> ; wr[k,h] = <Wdst[k,h,:], attn_r[h,:]>
    headdot768_kernel<<<(N_API * NH + 7) / 8, 256, 0, stream>>>(Fsrc, attn_l, el, N_API);
    headdot768_kernel<<<(NINP  * NH + 7) / 8, 256, 0, stream>>>(Wdst, attn_r, wr, NINP);

    // 5) er = Xfile @ wr  (folds fdst away)
    er_kernel<<<(N_FILE * NH + 7) / 8, 256, 0, stream>>>(Xfile, wr, er, N_FILE);

    // 6) edge softmax
    init_mz_kernel<<<(N_FILE * NH + 255) / 256, 256, 0, stream>>>(mbuf, zbuf, N_FILE * NH);
    edge_e_kernel<<<(NEDGE * NH + 255) / 256, 256, 0, stream>>>(src, dst, el, er, ebuf, mbuf);
    edge_a_kernel<<<(NEDGE * NH + 255) / 256, 256, 0, stream>>>(dst, ebuf, mbuf, zbuf);

    // 7) aggregation: rst = gat_bias + sum_e alpha * Fsrc[src]
    rst_init_kernel<<<(N_FILE * FDIM + 255) / 256, 256, 0, stream>>>(rst, gat_bias, N_FILE * FDIM);
    agg_kernel<<<NEDGE, 256, 0, stream>>>(src, dst, ebuf, zbuf, Fsrc, rst);

    // 8) out = rst @ Whead + bhead
    gemm_wmma<false, false><<<dim3(NOUT / 128, (N_FILE + 63) / 64), 256, 0, stream>>>(
        rst, WheadT, bhead, out, N_FILE, NOUT, FDIM);
}
